// AttLayer_34660386079090
// MI455X (gfx1250) — compile-verified, matrix operand order checked by reading, not run on hardware
//
#include <hip/hip_runtime.h>
#include <hip/hip_bf16.h>
#include <math.h>
#include <stdint.h>

typedef __bf16 bf16;
typedef __attribute__((ext_vector_type(16))) __bf16 v16bf;
typedef __attribute__((ext_vector_type(8)))  __bf16 v8bf;
typedef __attribute__((ext_vector_type(8)))  float  v8f;

#define BB 8
#define DD 512
#define LL 2048
#define CC 256

union Frag { v16bf v; v8bf h[2]; };

// CDNA5 bf16 A/B fragment layout (16x32 / 32x16):
// lane<16 holds K = {0..7, 16..23}, lane>=16 holds K = {8..15, 24..31}.
// Caller passes pointer already offset by khalf = (lane&16)?8:0.
__device__ __forceinline__ v16bf load_frag(const bf16* p) {
    Frag f;
    f.h[0] = *(const v8bf*)(p);
    f.h[1] = *(const v8bf*)(p + 16);
    return f.v;
}

__device__ __forceinline__ v8f wmma_bf16(v16bf a, v16bf b, v8f c) {
    return __builtin_amdgcn_wmma_f32_16x16x32_bf16(false, a, false, b,
                                                   (short)0, c, false, false);
}

// Async global->LDS 16-byte copy (CDNA5, tracked by ASYNCcnt).
__device__ __forceinline__ void async_copy_b128(uint32_t lds_addr,
                                                const void* gptr) {
    asm volatile("global_load_async_to_lds_b128 %0, %1, off"
                 :: "v"(lds_addr), "v"(gptr) : "memory");
}
__device__ __forceinline__ void wait_asynccnt0() {
    asm volatile("s_wait_asynccnt 0x0" ::: "memory");
}

// ---------------------------------------------------------------------------
// Kernel 0: fp32 -> bf16 weight conversion
// ---------------------------------------------------------------------------
__global__ void cvt_bf16_kernel(const float* __restrict__ src,
                                bf16* __restrict__ dst, int n) {
    int i = blockIdx.x * blockDim.x + threadIdx.x;
    if (i < n) dst[i] = (bf16)src[i];
}

// ---------------------------------------------------------------------------
// Kernel 1: QKV projection.  q,k,v = W(C,D) x x1(D,L) + bias.
// Block = 256 thr (8 waves, 2x4), tile 64(c) x 128(l), K-loop over D.
// x1 tile is staged to LDS transposed (bf16, K=d contiguous).
// Stores: q,k -> [b][l][c] (bf16), v -> [b][c][l] (bf16).
// ---------------------------------------------------------------------------
__global__ __launch_bounds__(256) void qkv_kernel(
    const float* __restrict__ x1,
    const bf16* __restrict__ Wq, const bf16* __restrict__ Wk,
    const bf16* __restrict__ Wv,
    const float* __restrict__ bq, const float* __restrict__ bk,
    const float* __restrict__ bv,
    bf16* __restrict__ qo, bf16* __restrict__ ko, bf16* __restrict__ vo) {
    __shared__ bf16 Bt[128][40];   // [l][d] tile, padded, 16B-aligned rows

    const int b  = blockIdx.z;
    const int c0 = blockIdx.y * 64;
    const int l0 = blockIdx.x * 128;
    const int tid = threadIdx.x;
    const int lane = tid & 31;
    const int w = tid >> 5;
    const int wc = w & 1;          // 2 waves along c
    const int wl = w >> 1;         // 4 waves along l
    const int cbase = c0 + wc * 32;
    const int lbase = l0 + wl * 32;
    const int n  = lane & 15;
    const int kh = (lane & 16) >> 1;   // 0 or 8
    const int hi = (lane & 16) ? 8 : 0;

    v8f aq[2][2] = {}, ak[2][2] = {}, av[2][2] = {};

    for (int kk = 0; kk < DD; kk += 32) {
        __syncthreads();
        // Stage x1[kk..kk+31][l0..l0+127] -> Bt[l][d] as bf16 (transpose+cvt)
        #pragma unroll
        for (int i = 0; i < 16; ++i) {
            int idx = tid + i * 256;
            int d = idx >> 7;       // 0..31
            int l = idx & 127;      // 0..127
            float xv = x1[((size_t)b * DD + (kk + d)) * LL + l0 + l];
            Bt[l][d] = (bf16)xv;
        }
        __syncthreads();

        v16bf bfr[2];
        #pragma unroll
        for (int ln = 0; ln < 2; ++ln) {
            int col = wl * 32 + ln * 16 + n;
            bfr[ln] = load_frag(&Bt[col][kh]);
        }
        #pragma unroll
        for (int cm = 0; cm < 2; ++cm) {
            const size_t ao = (size_t)(cbase + cm * 16 + n) * DD + kk + kh;
            v16bf fq = load_frag(Wq + ao);
            v16bf fk = load_frag(Wk + ao);
            v16bf fv = load_frag(Wv + ao);
            #pragma unroll
            for (int ln = 0; ln < 2; ++ln) {
                aq[cm][ln] = wmma_bf16(fq, bfr[ln], aq[cm][ln]);
                ak[cm][ln] = wmma_bf16(fk, bfr[ln], ak[cm][ln]);
                av[cm][ln] = wmma_bf16(fv, bfr[ln], av[cm][ln]);
            }
        }
    }

    // Epilogue: acc element (VGPR r) is row M=r+hi (c), col N=lane&15 (l)
    #pragma unroll
    for (int cm = 0; cm < 2; ++cm) {
        #pragma unroll
        for (int ln = 0; ln < 2; ++ln) {
            #pragma unroll
            for (int r = 0; r < 8; ++r) {
                int c = cbase + cm * 16 + r + hi;
                int l = lbase + ln * 16 + n;
                float fq = aq[cm][ln][r] + bq[c];
                float fk = ak[cm][ln][r] + bk[c];
                float fv = av[cm][ln][r] + bv[c];
                qo[((size_t)b * LL + l) * CC + c] = (bf16)fq;
                ko[((size_t)b * LL + l) * CC + c] = (bf16)fk;
                vo[((size_t)b * CC + c) * LL + l] = (bf16)fv;
            }
        }
    }
}

// ---------------------------------------------------------------------------
// Kernel 2: fused scores + softmax for 16 rows of att.
// att[l][m] = softmax_m( q^T k / sqrt(C) + log(mask+1e-6) ) * mask
// Block = 256 thr; q fragments loaded once (K=C=256 -> 8 frags), scores in
// 128 KB LDS, per-row reduction via 16-lane shuffles, bf16 out [b][l][m].
// ---------------------------------------------------------------------------
__global__ __launch_bounds__(256) void attn_softmax_kernel(
    const bf16* __restrict__ qb, const bf16* __restrict__ kb,
    const float* __restrict__ mask, bf16* __restrict__ attb) {
    extern __shared__ float S[];   // 16 * 2048 floats

    const int b  = blockIdx.y;
    const int l0 = blockIdx.x * 16;
    const int tid = threadIdx.x;
    const int lane = tid & 31;
    const int w = tid >> 5;
    const int n  = lane & 15;
    const int kh = (lane & 16) >> 1;
    const int hi = (lane & 16) ? 8 : 0;

    // A fragments: q rows l0..l0+15, all of K (C=256)
    v16bf afr[8];
    {
        const size_t qrow = ((size_t)b * LL + l0 + n) * CC;
        #pragma unroll
        for (int kk = 0; kk < 8; ++kk)
            afr[kk] = load_frag(qb + qrow + kk * 32 + kh);
    }

    // Sweep columns: 8 waves x 16 cols = 128 per iter, 16 iters
    for (int it = 0; it < 16; ++it) {
        const int m0 = it * 128 + w * 16;
        v8f acc = {};
        const size_t krow = ((size_t)b * LL + m0 + n) * CC;
        // prefetch next iteration's k row for this lane
        if (it < 15)
            __builtin_prefetch(kb + krow + 128 * CC, 0, 1);
        #pragma unroll
        for (int kk = 0; kk < 8; ++kk) {
            v16bf bfr = load_frag(kb + krow + kk * 32 + kh);
            acc = wmma_bf16(afr[kk], bfr, acc);
        }
        #pragma unroll
        for (int r = 0; r < 8; ++r)
            S[(r + hi) * LL + m0 + n] = acc[r];
    }
    __syncthreads();

    // Softmax: wave w owns rows 2w (lanes 0-15) and 2w+1 (lanes 16-31)
    const int row = 2 * w + ((lane & 16) ? 1 : 0);
    const float* pm = mask + (size_t)b * LL;
    const float scale = 0.0625f;   // 1/sqrt(256)
    float* srow = S + (size_t)row * LL;

    float mx = -1e30f;
    for (int j = n; j < LL; j += 16) {
        float v = srow[j] * scale + __logf(pm[j] + 1e-6f);
        srow[j] = v;
        mx = fmaxf(mx, v);
    }
    #pragma unroll
    for (int d = 1; d < 16; d <<= 1) mx = fmaxf(mx, __shfl_xor(mx, d, 32));

    float sum = 0.0f;
    for (int j = n; j < LL; j += 16) {
        float e = __expf(srow[j] - mx);
        srow[j] = e;
        sum += e;
    }
    #pragma unroll
    for (int d = 1; d < 16; d <<= 1) sum += __shfl_xor(sum, d, 32);
    const float inv = 1.0f / sum;

    bf16* arow = attb + ((size_t)b * LL + l0 + row) * LL;
    for (int j = n; j < LL; j += 16)
        arow[j] = (bf16)(srow[j] * inv * pm[j]);
}

// ---------------------------------------------------------------------------
// Kernel 3: out = relu( v(C,L) x att^T(L,L) ), K = m = 2048.
// Double-buffered GLOBAL_LOAD_ASYNC_TO_LDS_B128 staging of both tiles
// (ASYNCcnt pipeline): issue K-step i+1 copies while WMMA-ing K-step i.
// Store bf16 [b][l][c].
// ---------------------------------------------------------------------------
__global__ __launch_bounds__(256) void av_kernel(
    const bf16* __restrict__ vb, const bf16* __restrict__ attb,
    bf16* __restrict__ ob) {
    __shared__ bf16 Vt[2][64][40];    // v tile   [c][k], padded rows (80 B)
    __shared__ bf16 At[2][128][40];   // att tile [l][k], padded rows (80 B)

    const int b  = blockIdx.z;
    const int c0 = blockIdx.y * 64;
    const int l0 = blockIdx.x * 128;
    const int tid = threadIdx.x;
    const int lane = tid & 31;
    const int w = tid >> 5;
    const int wc = w & 1;
    const int wl = w >> 1;
    const int n  = lane & 15;
    const int kh = (lane & 16) >> 1;
    const int hi = (lane & 16) ? 8 : 0;

    const bf16* vbase = vb + ((size_t)b * CC + c0) * LL;
    const bf16* abase = attb + ((size_t)b * LL + l0) * LL;

    // Issue async copies of one K-step (32 wide) into buffer ib.
    // 64*4 + 128*4 = 768 b128 copies = 3 per thread.
    auto stage = [&](int kk, int ib) {
        #pragma unroll
        for (int t = 0; t < 3; ++t) {
            int q = tid + t * 256;
            if (q < 256) {
                int row = q >> 2, seg = q & 3;
                uint32_t lds = (uint32_t)(uintptr_t)&Vt[ib][row][seg * 8];
                const bf16* g = vbase + (size_t)row * LL + kk + seg * 8;
                async_copy_b128(lds, g);
            } else {
                int idx = q - 256;
                int row = idx >> 2, seg = idx & 3;
                uint32_t lds = (uint32_t)(uintptr_t)&At[ib][row][seg * 8];
                const bf16* g = abase + (size_t)row * LL + kk + seg * 8;
                async_copy_b128(lds, g);
            }
        }
    };

    v8f acc[2][2] = {};

    stage(0, 0);
    wait_asynccnt0();
    __syncthreads();

    for (int kk = 0; kk < LL; kk += 32) {
        const int ib = (kk >> 5) & 1;
        if (kk + 32 < LL) stage(kk + 32, ib ^ 1);   // overlap with compute

        v16bf afr[2], bfr[2];
        #pragma unroll
        for (int cm = 0; cm < 2; ++cm)
            afr[cm] = load_frag(&Vt[ib][wc * 32 + cm * 16 + n][kh]);
        #pragma unroll
        for (int ln = 0; ln < 2; ++ln)
            bfr[ln] = load_frag(&At[ib][wl * 32 + ln * 16 + n][kh]);
        #pragma unroll
        for (int cm = 0; cm < 2; ++cm)
            #pragma unroll
            for (int ln = 0; ln < 2; ++ln)
                acc[cm][ln] = wmma_bf16(afr[cm], bfr[ln], acc[cm][ln]);

        wait_asynccnt0();      // my next-buffer copies have landed in LDS
        __syncthreads();       // everyone's copies landed & reads done
    }

    #pragma unroll
    for (int cm = 0; cm < 2; ++cm)
        #pragma unroll
        for (int ln = 0; ln < 2; ++ln)
            #pragma unroll
            for (int r = 0; r < 8; ++r) {
                int c = c0 + wc * 32 + cm * 16 + r + hi;
                int l = l0 + wl * 32 + ln * 16 + n;
                float x = acc[cm][ln][r];
                ob[((size_t)b * LL + l) * CC + c] = (bf16)fmaxf(x, 0.0f);
            }
}

// ---------------------------------------------------------------------------
// Kernel 4: final projection out = Wo(D,C) x relu_out + bo, * mask. K = C.
// fp32 store to d_out [b][d][l].
// ---------------------------------------------------------------------------
__global__ __launch_bounds__(256) void oproj_kernel(
    const bf16* __restrict__ Wo, const bf16* __restrict__ ob,
    const float* __restrict__ bo, const float* __restrict__ mask,
    float* __restrict__ out) {
    const int b  = blockIdx.z;
    const int d0 = blockIdx.y * 64;
    const int l0 = blockIdx.x * 128;
    const int tid = threadIdx.x;
    const int lane = tid & 31;
    const int w = tid >> 5;
    const int dbase = d0 + (w & 1) * 32;
    const int lbase = l0 + (w >> 1) * 32;
    const int n  = lane & 15;
    const int kh = (lane & 16) >> 1;
    const int hi = (lane & 16) ? 8 : 0;

    v8f acc[2][2] = {};
    const size_t arow0 = (size_t)(dbase + n) * CC;
    const size_t brow0 = ((size_t)b * LL + lbase + n) * CC;

    #pragma unroll
    for (int kk = 0; kk < CC; kk += 32) {
        v16bf afr[2], bfr[2];
        #pragma unroll
        for (int dm = 0; dm < 2; ++dm)
            afr[dm] = load_frag(Wo + arow0 + (size_t)dm * 16 * CC + kk + kh);
        #pragma unroll
        for (int ln = 0; ln < 2; ++ln)
            bfr[ln] = load_frag(ob + brow0 + (size_t)ln * 16 * CC + kk + kh);
        #pragma unroll
        for (int dm = 0; dm < 2; ++dm)
            #pragma unroll
            for (int ln = 0; ln < 2; ++ln)
                acc[dm][ln] = wmma_bf16(afr[dm], bfr[ln], acc[dm][ln]);
    }

    #pragma unroll
    for (int dm = 0; dm < 2; ++dm)
        #pragma unroll
        for (int ln = 0; ln < 2; ++ln)
            #pragma unroll
            for (int r = 0; r < 8; ++r) {
                int d = dbase + dm * 16 + r + hi;
                int l = lbase + ln * 16 + n;
                float x = acc[dm][ln][r] + bo[d];
                x *= mask[(size_t)b * LL + l];
                out[((size_t)b * DD + d) * LL + l] = x;
            }
}

// ---------------------------------------------------------------------------
extern "C" void kernel_launch(void* const* d_in, const int* in_sizes, int n_in,
                              void* d_out, int out_size, void* d_ws,
                              size_t ws_size, hipStream_t stream) {
    (void)in_sizes; (void)n_in; (void)out_size; (void)ws_size;
    const float* x1   = (const float*)d_in[0];
    // d_in[1] (x2) is unused by the reference
    const float* mask = (const float*)d_in[2];
    const float* Wq   = (const float*)d_in[3];
    const float* bq   = (const float*)d_in[4];
    const float* Wk   = (const float*)d_in[5];
    const float* bk   = (const float*)d_in[6];
    const float* Wv   = (const float*)d_in[7];
    const float* bv   = (const float*)d_in[8];
    const float* Wo   = (const float*)d_in[9];
    const float* bo   = (const float*)d_in[10];
    float* out = (float*)d_out;

    const size_t WSZ = (size_t)CC * DD;          // 131072 weight elems
    char* p = (char*)d_ws;
    bf16* Wqb = (bf16*)p; p += WSZ * 2;
    bf16* Wkb = (bf16*)p; p += WSZ * 2;
    bf16* Wvb = (bf16*)p; p += WSZ * 2;
    bf16* Wob = (bf16*)p; p += WSZ * 2;
    bf16* qb  = (bf16*)p; p += (size_t)BB * LL * CC * 2;
    bf16* kb  = (bf16*)p; p += (size_t)BB * LL * CC * 2;
    bf16* vb  = (bf16*)p; p += (size_t)BB * CC * LL * 2;
    bf16* ob  = (bf16*)p; p += (size_t)BB * LL * CC * 2;
    bf16* attb = (bf16*)p;                       // B*L*L bf16 = 64 MiB

    // 0) weight conversion
    {
        dim3 g((unsigned)((WSZ + 255) / 256));
        cvt_bf16_kernel<<<g, 256, 0, stream>>>(Wq, Wqb, (int)WSZ);
        cvt_bf16_kernel<<<g, 256, 0, stream>>>(Wk, Wkb, (int)WSZ);
        cvt_bf16_kernel<<<g, 256, 0, stream>>>(Wv, Wvb, (int)WSZ);
        cvt_bf16_kernel<<<g, 256, 0, stream>>>(Wo, Wob, (int)WSZ);
    }
    // 1) QKV projection
    qkv_kernel<<<dim3(LL / 128, CC / 64, BB), 256, 0, stream>>>(
        x1, Wqb, Wkb, Wvb, bq, bk, bv, qb, kb, vb);
    // 2) scores + softmax (128 KB dynamic LDS per block)
    attn_softmax_kernel<<<dim3(LL / 16, BB), 256, 16 * LL * sizeof(float),
                          stream>>>(qb, kb, mask, attb);
    // 3) att x V + relu (async-LDS double-buffered)
    av_kernel<<<dim3(LL / 128, CC / 64, BB), 256, 0, stream>>>(vb, attb, ob);
    // 4) output projection + bias + mask
    oproj_kernel<<<dim3(LL / 128, DD / 64, BB), 256, 0, stream>>>(
        Wob, ob, bo, mask, out);
}
